// ExtSummarizer_46548855554801
// MI455X (gfx1250) — matrix operand order repaired and checked by reading, not compiled, not run on hardware
//
#include <hip/hip_runtime.h>
#include <stdint.h>

// ---------------------------------------------------------------------------
// ExtSummarizer cal_matrix0, batched.  B=1024, N=50, H=768.
// Dominant cost: T = S @ W_sim  ([51200,768]x[768,768], ~60 GFLOP) -> WMMA bf16.
// The 50x50 linear solve is rank-1 (Sherman-Morrison closed form).
// ---------------------------------------------------------------------------

#define Hh   768
#define Nn   50
#define MP   64          // padded sentence rows (4 x 16)
#define Bb   1024
#define LAMB 0.8f

typedef __attribute__((ext_vector_type(16))) __bf16 bf16x16;
typedef __attribute__((ext_vector_type(8)))  float  f32x8;

union Frag {
    bf16x16 v;
    uint4   q[2];
};

// ---------------- LDS layout for main kernel (bytes) -----------------------
#define SB_ST   776                    // bf16 row stride (768 + 8 pad -> bank spread)
#define QS      68                     // f32 qmat row stride
#define OFF_SB  0                      // bf16 [64][776]  = 99328
#define OFF_TB  99328                  // bf16 [64][776]  = 99328
#define OFF_Q   198656                 // f32  [64][68]   = 17408
#define OFF_RC  216064                 // f32  [64][8]    = 2048
#define OFF_RR  218112                 // f32  [64][8]    = 2048
#define OFF_CT  220160                 // f32  [64]
#define OFF_RL  220416                 // f32  [64]
#define OFF_SV  220672                 // f32  [64]
#define OFF_UV  220928                 // f32  [64]
#define OFF_MK  221184                 // f32  [64]
#define OFF_SC  221440                 // f32  [8]
#define SMEM_BYTES 221472

__device__ __forceinline__ float sigmoidf_(float x) {
    return 1.0f / (1.0f + __expf(-x));
}

// ---------------------------------------------------------------------------
// K0: per-batch mean of sentence vectors: D[b][h] = mean_n S[b][n][h]
// ---------------------------------------------------------------------------
__global__ __launch_bounds__(256) void dmean_kernel(const float* __restrict__ X,
                                                    float* __restrict__ D) {
    int b = blockIdx.x;
    int h = blockIdx.y * 256 + threadIdx.x;          // gridDim.y == 3 -> h < 768
    const float* p = X + (size_t)b * Nn * Hh + h;
    float s = 0.f;
    #pragma unroll
    for (int n = 0; n < Nn; ++n) s += p[n * Hh];
    D[(size_t)b * Hh + h] = s * (1.0f / (float)Nn);
}

// ---------------------------------------------------------------------------
// K1: pack W_sim (f32, row-major [h][col]) into bf16 WMMA B-fragment order.
// Layout: Wp[((kt*48 + nt)*32 + lane)*16 + i], i = consecutive K within the
// lane's K-half:  K = kt*32 + (lane>=16 ? 16 : 0) + i,  col = nt*16 + lane%16.
// ---------------------------------------------------------------------------
__global__ __launch_bounds__(256) void packW_kernel(const float* __restrict__ W,
                                                    __bf16* __restrict__ Wp) {
    int idx = blockIdx.x * 256 + threadIdx.x;        // 24*48*32*16 = 589824
    int i    = idx & 15;
    int r    = idx >> 4;
    int lane = r & 31;  r >>= 5;
    int nt   = r % 48;
    int kt   = r / 48;
    int col  = nt * 16 + (lane & 15);
    int k    = kt * 32 + ((lane >> 4) * 16) + i;
    Wp[idx]  = (__bf16)W[(size_t)k * Hh + col];
}

// ---------------------------------------------------------------------------
// K2: R = D @ W_rel^T  ([1024,768]x[768,768]^T), WMMA, on-the-fly bf16 cvt.
// One 16x16 tile per wave; 64x48 tiles; 8 waves / block -> 384 blocks.
// ---------------------------------------------------------------------------
__global__ __launch_bounds__(256) void rgemm_kernel(const float* __restrict__ D,
                                                    const float* __restrict__ Wrel,
                                                    float* __restrict__ R) {
    int wave  = blockIdx.x * 8 + (threadIdx.x >> 5);
    int lane  = threadIdx.x & 31;
    int mt    = wave / 48;                           // batch tile (0..63)
    int nt    = wave % 48;                           // h tile (0..47)
    int brow  = mt * 16 + (lane & 15);
    int hrow  = nt * 16 + (lane & 15);
    int kb    = (lane >> 4) * 16;                    // A K-half base
    int khalf = (lane >> 4) * 16;                    // B K-half base
    f32x8 acc = {};
    for (int ks = 0; ks < 24; ++ks) {
        Frag a, bb;
        const float* dp = D + (size_t)brow * Hh + ks * 32;
        #pragma unroll
        for (int i = 0; i < 8; ++i) a.v[i]     = (__bf16)dp[kb + i];
        #pragma unroll
        for (int i = 0; i < 8; ++i) a.v[8 + i] = (__bf16)dp[kb + 8 + i];
        const float* wp = Wrel + (size_t)hrow * Hh + ks * 32 + khalf;
        #pragma unroll
        for (int i = 0; i < 16; ++i) bb.v[i] = (__bf16)wp[i];
        acc = __builtin_amdgcn_wmma_f32_16x16x32_bf16(false, a.v, false, bb.v,
                                                      (short)0, acc, false, false);
    }
    int mloc = (lane >> 4) * 8;
    int nloc = lane & 15;
    #pragma unroll
    for (int r2 = 0; r2 < 8; ++r2)
        R[(size_t)(mt * 16 + mloc + r2) * Hh + nt * 16 + nloc] = acc[r2];
}

// ---------------------------------------------------------------------------
// K3: per-batch fused pipeline.  One block (512 thr = 16 waves) per batch item.
// ---------------------------------------------------------------------------
__global__ __launch_bounds__(512) void main_kernel(
        const float* __restrict__ X, const unsigned char* __restrict__ mask,
        const float* __restrict__ Wc, const __bf16* __restrict__ Wp,
        const float* __restrict__ R, const float* __restrict__ bmat,
        const float* __restrict__ qw, const float* __restrict__ qb,
        float* __restrict__ out) {
    extern __shared__ char smem[];
    __bf16* Sb   = (__bf16*)(smem + OFF_SB);
    __bf16* Tb   = (__bf16*)(smem + OFF_TB);
    float*  qmat = (float*)(smem + OFF_Q);
    float*  redC = (float*)(smem + OFF_RC);
    float*  redR = (float*)(smem + OFF_RR);
    float*  contv= (float*)(smem + OFF_CT);
    float*  relv = (float*)(smem + OFF_RL);
    float*  sv   = (float*)(smem + OFF_SV);
    float*  uv   = (float*)(smem + OFF_UV);
    float*  mk   = (float*)(smem + OFF_MK);
    float*  scal = (float*)(smem + OFF_SC);

    const int b    = blockIdx.x;
    const int t    = threadIdx.x;
    const int lane = t & 31;
    const int wave = t >> 5;

    // ---- phase 0: S_b -> LDS bf16, zero pad rows 50..63 --------------------
    const float* Xb = X + (size_t)b * Nn * Hh;
    for (int idx = t; idx < Nn * Hh; idx += 512) {
        int row = idx / Hh, col = idx - row * Hh;
        Sb[row * SB_ST + col] = (__bf16)Xb[idx];
    }
    for (int idx = Nn * SB_ST + t; idx < MP * SB_ST; idx += 512)
        Sb[idx] = (__bf16)0.0f;
    __syncthreads();

    // ---- phase 1: T = S @ W_sim  (64x768, K=768), 12 acc tiles / wave ------
    {
        const int rlo = lane & 15;
        const int kb  = (lane >> 4) * 16;
        f32x8 acc[4][3] = {};
        for (int ks = 0; ks < 24; ++ks) {
            Frag a[4];
            #pragma unroll
            for (int mt = 0; mt < 4; ++mt) {
                const __bf16* p = Sb + (mt * 16 + rlo) * SB_ST + ks * 32;
                a[mt].q[0] = *(const uint4*)(p + kb);
                a[mt].q[1] = *(const uint4*)(p + kb + 8);
            }
            #pragma unroll
            for (int j = 0; j < 3; ++j) {
                int nt = wave * 3 + j;
                Frag bb;
                const __bf16* wp2 = Wp + ((size_t)((ks * 48 + nt) * 32 + lane)) * 16;
                bb.q[0] = *(const uint4*)(wp2);
                bb.q[1] = *(const uint4*)(wp2 + 8);
                #pragma unroll
                for (int mt = 0; mt < 4; ++mt)
                    acc[mt][j] = __builtin_amdgcn_wmma_f32_16x16x32_bf16(
                        false, a[mt].v, false, bb.v, (short)0, acc[mt][j], false, false);
            }
        }
        const int mloc = (lane >> 4) * 8, nloc = lane & 15;
        #pragma unroll
        for (int mt = 0; mt < 4; ++mt)
            #pragma unroll
            for (int j = 0; j < 3; ++j) {
                int nt = wave * 3 + j;
                #pragma unroll
                for (int r2 = 0; r2 < 8; ++r2)
                    Tb[(mt * 16 + mloc + r2) * SB_ST + nt * 16 + nloc] =
                        (__bf16)acc[mt][j][r2];
            }
    }
    __syncthreads();

    // ---- phase 2: sc_sim = T @ S^T  (64x64, K=768), 1 tile / wave ----------
    {
        const int mt = wave >> 2, nt = wave & 3;
        const int rlo = lane & 15;
        const int kb = (lane >> 4) * 16, khalf = (lane >> 4) * 16;
        f32x8 acc2 = {};
        for (int ks = 0; ks < 24; ++ks) {
            Frag a, bb;
            const __bf16* tp = Tb + (mt * 16 + rlo) * SB_ST + ks * 32;
            a.q[0] = *(const uint4*)(tp + kb);
            a.q[1] = *(const uint4*)(tp + kb + 8);
            const __bf16* sp = Sb + (nt * 16 + rlo) * SB_ST + ks * 32 + khalf;
            bb.q[0] = *(const uint4*)(sp);
            bb.q[1] = *(const uint4*)(sp + 8);
            acc2 = __builtin_amdgcn_wmma_f32_16x16x32_bf16(false, a.v, false, bb.v,
                                                           (short)0, acc2, false, false);
        }
        const int mloc = (lane >> 4) * 8, nloc = lane & 15;
        #pragma unroll
        for (int r2 = 0; r2 < 8; ++r2)
            qmat[(mt * 16 + mloc + r2) * QS + nt * 16 + nloc] = acc2[r2];
    }
    __syncthreads();

    // ---- phase 3: sc_cont / sc_rel dot products (8 partials per row) -------
    {
        int n = t >> 3, p = t & 7;
        const float* Rb = R + (size_t)b * Hh;
        const __bf16* srow = Sb + n * SB_ST + p * 96;
        const float* wc = Wc + p * 96;
        const float* rr = Rb + p * 96;
        float pc = 0.f, pr = 0.f;
        for (int h = 0; h < 96; ++h) {
            float sx = (float)srow[h];
            pc = fmaf(wc[h], sx, pc);
            pr = fmaf(rr[h], sx, pr);
        }
        redC[n * 8 + p] = pc;
        redR[n * 8 + p] = pr;
    }
    __syncthreads();
    if (t < 64) {
        float c0 = 0.f, r0 = 0.f;
        #pragma unroll
        for (int p = 0; p < 8; ++p) { c0 += redC[t * 8 + p]; r0 += redR[t * 8 + p]; }
        contv[t] = c0; relv[t] = r0;
    }
    if (t < Nn) mk[t] = mask[(size_t)b * Nn + t] ? 1.0f : 0.0f;
    __syncthreads();

    // ---- phase 4: s[n] = sum_m q[n,m]*q_w[m] + q_b -------------------------
    if (t < Nn) {
        float base = relv[t] + contv[t] + bmat[0];
        float mn = mk[t];
        float s = qb[0];
        for (int m = 0; m < Nn; ++m)
            s = fmaf((base + qmat[t * QS + m]) * (mn * mk[m]), qw[m], s);
        sv[t] = s;
    }
    __syncthreads();

    // ---- phase 5: layernorm over N, sigmoid, rank-1 closed form ------------
    if (t == 0) {
        float sn = 0.f;
        for (int n = 0; n < Nn; ++n) sn += mk[n];
        float mu = 0.f;
        for (int n = 0; n < Nn; ++n) mu += sv[n];
        mu *= (1.0f / (float)Nn);
        float var = 0.f;
        for (int n = 0; n < Nn; ++n) { float d = sv[n] - mu; var = fmaf(d, d, var); }
        var *= (1.0f / (float)Nn);
        float rinv = rsqrtf(var + 1e-6f);
        float c = 0.f;
        for (int n = 0; n < Nn; ++n) {
            float u = mk[n] * sigmoidf_((sv[n] - mu) * rinv);
            uv[n] = u; c += u;
        }
        scal[0] = sn;
        scal[1] = fmaxf(c, 1e-12f);
    }
    __syncthreads();
    if (t < Nn) {
        float sn = scal[0], c = scal[1];
        // (I - lamb*u*w^T)^-1 closed form:  x = (1-lamb)/sn + lamb*u/c
        float x = (1.0f - LAMB) / sn + LAMB * uv[t] / c;
        out[(size_t)b * Nn + t] = mk[t] * sigmoidf_(x * sn);
    }
}

// ---------------------------------------------------------------------------
extern "C" void kernel_launch(void* const* d_in, const int* in_sizes, int n_in,
                              void* d_out, int out_size, void* d_ws, size_t ws_size,
                              hipStream_t stream) {
    (void)in_sizes; (void)n_in; (void)out_size; (void)ws_size;
    const float*         X    = (const float*)d_in[0];
    const unsigned char* mask = (const unsigned char*)d_in[1];
    const float*         Wc   = (const float*)d_in[2];
    const float*         Wsim = (const float*)d_in[3];
    const float*         Wrel = (const float*)d_in[4];
    const float*         bmat = (const float*)d_in[5];
    const float*         qwv  = (const float*)d_in[6];
    const float*         qbv  = (const float*)d_in[7];
    float*               out  = (float*)d_out;

    char* ws = (char*)d_ws;
    __bf16* Wp = (__bf16*)ws;                              // 589824 bf16 = 1,179,648 B
    float*  D  = (float*)(ws + 1179648);                   // 1024*768 f32 = 3,145,728 B
    float*  R  = (float*)(ws + 1179648 + 3145728);         // 1024*768 f32

    packW_kernel<<<2304, 256, 0, stream>>>(Wsim, Wp);
    dmean_kernel<<<dim3(Bb, 3), 256, 0, stream>>>(X, D);
    rgemm_kernel<<<384, 256, 0, stream>>>(D, Wrel, R);

    (void)hipFuncSetAttribute((const void*)main_kernel,
                              hipFuncAttributeMaxDynamicSharedMemorySize, SMEM_BYTES);
    main_kernel<<<Bb, 512, SMEM_BYTES, stream>>>(X, mask, Wc, Wp, R,
                                                 bmat, qwv, qbv, out);
}
